// MultiheadAttention_75393855914273
// MI455X (gfx1250) — matrix operand order repaired and checked by reading, not compiled, not run on hardware
//
#include <hip/hip_runtime.h>

// ---------------------------------------------------------------------------
// MI455X (gfx1250) multi-head self-attention, bf16 WMMA + TDM pipeline.
// ---------------------------------------------------------------------------

typedef __attribute__((ext_vector_type(16))) __bf16 v16bf;
typedef __attribute__((ext_vector_type(8)))  float  v8f;
typedef __attribute__((ext_vector_type(4)))  unsigned int v4u;
typedef __attribute__((ext_vector_type(4)))  int v4i;
typedef __attribute__((ext_vector_type(8)))  int v8i;

#define T_SEQ  2048
#define BATCH  2
#define EMB    1024
#define NH     16
#define HD     64
#define M_ROWS (T_SEQ * BATCH)   // 4096
#define N3     (3 * EMB)         // 3072
#define BHEADS (BATCH * NH)      // 32
#define SCALE  0.125f            // HD^-0.5

// ---------------------------------------------------------------------------
// WMMA wrapper: D = A(16x32 bf16) * B(32x16 bf16) + C(16x16 f32)
// ---------------------------------------------------------------------------
__device__ __forceinline__ v8f wmma_bf16(v16bf a, v16bf b, v8f c) {
  return __builtin_amdgcn_wmma_f32_16x16x32_bf16(
      /*neg_a=*/false, a, /*neg_b=*/false, b,
      /*c_mod=*/(short)0, c, /*reuse_a=*/false, /*reuse_b=*/false);
}

// A fragment (16x32, row-major source, leading dim ld).
// lane L: M = L&15 ; element i -> K = (L>>4)*16 + i (contiguous per lane)
__device__ __forceinline__ v16bf load_a(const __bf16* p, int ld) {
  int lane = threadIdx.x & 31;
  const __bf16* row = p + (lane & 15) * ld + ((lane >> 4) << 4);
  v16bf a;
#pragma unroll
  for (int i = 0; i < 16; ++i)
    a[i] = row[i];
  return a;
}

// B fragment where B[k][n] = W[n][k] (B = W^T), W row-major, leading dim ld.
// lane L: N = L&15 ; element i -> K = (L>>4)*16 + i (contiguous per lane)
__device__ __forceinline__ v16bf load_bT(const __bf16* p, int ld) {
  int lane = threadIdx.x & 31;
  const __bf16* row = p + (lane & 15) * ld + ((lane >> 4) << 4);
  v16bf b;
#pragma unroll
  for (int i = 0; i < 16; ++i)
    b[i] = row[i];
  return b;
}

// B fragment from row-major B[k][n], leading dim ld.
__device__ __forceinline__ v16bf load_b(const __bf16* p, int ld) {
  int lane = threadIdx.x & 31;
  const __bf16* col = p + (((lane >> 4) << 4)) * ld + (lane & 15);
  v16bf b;
#pragma unroll
  for (int i = 0; i < 16; ++i)
    b[i] = col[i * ld];
  return b;
}

// ---------------------------------------------------------------------------
// Tensor Data Mover: async 2D tile (rows x 64 bf16, row stride HD) -> LDS.
// Descriptor per CDNA5 ISA ch.8: group0 {count, lds_addr, global_addr, type=2},
// group1 {data_size=2B, tensor_dim0=64, tensor_dim1, tile_dim0=64, tile_dim1,
//         tensor_dim0_stride=64}. Groups 2/3 zero (2D tensor).
// ---------------------------------------------------------------------------
__device__ __forceinline__ void tdm_load_tile(unsigned lds_off, const __bf16* gsrc,
                                              int rows, int tensor_rows) {
  unsigned long long ga = (unsigned long long)(size_t)gsrc;
  v4u g0 = { 1u,                                   // count=1 user descriptor
             lds_off,                              // lds_addr (bytes)
             (unsigned)(ga & 0xffffffffu),         // global_addr[31:0]
             (unsigned)((ga >> 32) & 0x1ffffffu) | (2u << 30) };  // type=2
  v8i g1 = { (int)(1u << 16),                      // data_size = 2 bytes
             (int)((unsigned)HD << 16),            // tensor_dim0 = 64
             (int)((unsigned)tensor_rows << 16),   // tensor_dim1
             (int)((unsigned)HD << 16),            // tile_dim0 = 64
             rows,                                 // tile_dim1
             HD,                                   // tensor_dim0_stride = 64
             0, 0 };
  v4i gz = { 0, 0, 0, 0 };
#if __has_builtin(__builtin_amdgcn_tensor_load_to_lds)
#if defined(__clang_major__) && (__clang_major__ >= 23)
  v8i gz8 = { 0, 0, 0, 0, 0, 0, 0, 0 };
  __builtin_amdgcn_tensor_load_to_lds(g0, g1, gz, gz, gz8, 0);
#else
  __builtin_amdgcn_tensor_load_to_lds(g0, g1, gz, gz, 0);
#endif
#else
  // Fallback: no copy is done (placeholder only), should not be hit on ROCm 7.2.
  __bf16* dst = (__bf16*)((char*)0 + lds_off);  // unused path placeholder
  (void)dst; (void)rows; (void)tensor_rows;
#endif
}

__device__ __forceinline__ void tdm_wait() {
#if __has_builtin(__builtin_amdgcn_s_wait_tensorcnt)
  __builtin_amdgcn_s_wait_tensorcnt(0);
#endif
}

// ---------------------------------------------------------------------------
// fp32 -> bf16 conversion (vectorized 4-wide)
// ---------------------------------------------------------------------------
__global__ void cvt_f32_bf16(const float* __restrict__ s, __bf16* __restrict__ d, int n4) {
  int i = blockIdx.x * blockDim.x + threadIdx.x;
  if (i < n4) {
    float4 f = ((const float4*)s)[i];
    d[i * 4 + 0] = (__bf16)f.x;
    d[i * 4 + 1] = (__bf16)f.y;
    d[i * 4 + 2] = (__bf16)f.z;
    d[i * 4 + 3] = (__bf16)f.w;
  }
}

// ---------------------------------------------------------------------------
// QKV projection: qkv[m,f] = X[m,:] . W[f,:] + bias[f]
// One wave per block computing a 32x64 output tile (8 WMMA per K step,
// B fragments reused across both A fragments).
// ---------------------------------------------------------------------------
__global__ void gemm_qkv(const __bf16* __restrict__ X, const __bf16* __restrict__ W,
                         const float* __restrict__ bias,
                         __bf16* __restrict__ Qb, __bf16* __restrict__ Kb,
                         __bf16* __restrict__ Vb) {
  int mt = blockIdx.x;   // 32-row tile of M
  int nt = blockIdx.y;   // 64-col tile of 3E
  int lane = threadIdx.x & 31;

  v8f acc[2][4] = {};
  const __bf16* xrow = X + (size_t)mt * 32 * EMB;
  for (int k0 = 0; k0 < EMB; k0 += 32) {
    if (k0 + 32 < EMB) {
      __builtin_prefetch(xrow + k0 + 32, 0, 0);
      __builtin_prefetch(xrow + 16 * EMB + k0 + 32, 0, 0);
    }
    v16bf a0 = load_a(xrow + k0, EMB);
    v16bf a1 = load_a(xrow + (size_t)16 * EMB + k0, EMB);
#pragma unroll
    for (int j = 0; j < 4; ++j) {
      v16bf b = load_bT(W + (size_t)(nt * 64 + j * 16) * EMB + k0, EMB);
      acc[0][j] = wmma_bf16(a0, b, acc[0][j]);
      acc[1][j] = wmma_bf16(a1, b, acc[1][j]);
    }
  }

#pragma unroll
  for (int u = 0; u < 2; ++u) {
#pragma unroll
    for (int j = 0; j < 4; ++j) {
      int f   = nt * 64 + j * 16 + (lane & 15);
      int sec = f >> 10;       // 0:q 1:k 2:v
      int ep  = f & 1023;
      int h   = ep >> 6;
      int d   = ep & 63;
      float bv = bias[f];
#pragma unroll
      for (int r = 0; r < 8; ++r) {
        int m = mt * 32 + u * 16 + ((lane >> 4) << 3) + r;
        int t = m >> 1;          // m = t*BATCH + b
        int bb = m & 1;
        float v = acc[u][j][r] + bv;
        size_t idx = ((size_t)(bb * NH + h) * T_SEQ + t) * HD + d;
        if (sec == 0)      Qb[idx] = (__bf16)(v * SCALE);
        else if (sec == 1) Kb[idx] = (__bf16)v;
        else               Vb[idx] = (__bf16)v;
      }
    }
  }
}

// ---------------------------------------------------------------------------
// Softmax pass 1: per-row running (max, sumexp) over all keys.
// grid = (T/16, BH); one wave per block.
// ---------------------------------------------------------------------------
__global__ void attn_stats(const __bf16* __restrict__ Qb, const __bf16* __restrict__ Kb,
                           float* __restrict__ mstat, float* __restrict__ lstat) {
  int qt = blockIdx.x;
  int bh = blockIdx.y;
  int lane = threadIdx.x & 31;

  const __bf16* q  = Qb + ((size_t)bh * T_SEQ + qt * 16) * HD;
  const __bf16* kp = Kb + (size_t)bh * T_SEQ * HD;

  v16bf a0 = load_a(q, HD);       // K = 0..31
  v16bf a1 = load_a(q + 32, HD);  // K = 32..63

  float rm[8], rs[8];
#pragma unroll
  for (int r = 0; r < 8; ++r) { rm[r] = -3.0e38f; rs[r] = 0.0f; }

  for (int s0 = 0; s0 < T_SEQ; s0 += 16) {
    v8f c = {};
    c = wmma_bf16(a0, load_bT(kp + (size_t)s0 * HD, HD), c);
    c = wmma_bf16(a1, load_bT(kp + (size_t)s0 * HD + 32, HD), c);
#pragma unroll
    for (int r = 0; r < 8; ++r) {
      float s  = c[r];
      float nm = fmaxf(rm[r], s);
      rs[r] = rs[r] * __expf(rm[r] - nm) + __expf(s - nm);
      rm[r] = nm;
    }
  }

  // merge across the 16 lanes that share each row (butterfly inside 16-group)
#pragma unroll
  for (int off = 1; off < 16; off <<= 1) {
#pragma unroll
    for (int r = 0; r < 8; ++r) {
      float om = __shfl_xor(rm[r], off, 32);
      float os = __shfl_xor(rs[r], off, 32);
      float nm = fmaxf(rm[r], om);
      rs[r] = rs[r] * __expf(rm[r] - nm) + os * __expf(om - nm);
      rm[r] = nm;
    }
  }

  if ((lane & 15) == 0) {
#pragma unroll
    for (int r = 0; r < 8; ++r) {
      int t = qt * 16 + ((lane >> 4) << 3) + r;
      mstat[(size_t)bh * T_SEQ + t] = rm[r];
      lstat[(size_t)bh * T_SEQ + t] = rs[r];
    }
  }
}

// ---------------------------------------------------------------------------
// Softmax pass 2 + PV: O = softmax(S) @ V, written to [T,B,E] bf16.
// K/V 32x64 tiles staged into LDS by the Tensor Data Mover (TENSORcnt),
// P tile relayout (C-frag -> A-frag) through LDS.
// grid = (T/16, BH); one wave per block; dynamic LDS:
//   [0,4K)   K tile   [4K,8K)  V tile   [8K,9K)  P tile
// ---------------------------------------------------------------------------
__global__ void attn_av(const __bf16* __restrict__ Qb, const __bf16* __restrict__ Kb,
                        const __bf16* __restrict__ Vb,
                        const float* __restrict__ mstat, const float* __restrict__ lstat,
                        __bf16* __restrict__ Ab) {
  extern __shared__ __align__(16) char smem[];
  __bf16* kt = (__bf16*)(smem);           // 32 x 64 bf16
  __bf16* vt = (__bf16*)(smem + 4096);    // 32 x 64 bf16
  __bf16* pt = (__bf16*)(smem + 8192);    // 16 x 32 bf16

  int qt = blockIdx.x;
  int bh = blockIdx.y;
  int lane = threadIdx.x & 31;

  const __bf16* q  = Qb + ((size_t)bh * T_SEQ + qt * 16) * HD;
  const __bf16* kp = Kb + (size_t)bh * T_SEQ * HD;
  const __bf16* vp = Vb + (size_t)bh * T_SEQ * HD;

  v16bf a0 = load_a(q, HD);
  v16bf a1 = load_a(q + 32, HD);

  float mrow[8], lrinv[8];
#pragma unroll
  for (int r = 0; r < 8; ++r) {
    int t = qt * 16 + ((lane >> 4) << 3) + r;
    mrow[r]  = mstat[(size_t)bh * T_SEQ + t];
    lrinv[r] = 1.0f / lstat[(size_t)bh * T_SEQ + t];
  }

  v8f o[4] = {};
  for (int s0 = 0; s0 < T_SEQ; s0 += 32) {
    // async DMA: next 32 keys / values into LDS
    tdm_load_tile(0u,    kp + (size_t)s0 * HD, 32, T_SEQ - s0);
    tdm_load_tile(4096u, vp + (size_t)s0 * HD, 32, T_SEQ - s0);
    tdm_wait();

#pragma unroll
    for (int half = 0; half < 2; ++half) {
      v8f c = {};
      c = wmma_bf16(a0, load_bT(kt + (half * 16) * HD, HD), c);
      c = wmma_bf16(a1, load_bT(kt + (half * 16) * HD + 32, HD), c);
#pragma unroll
      for (int r = 0; r < 8; ++r) {
        float p = __expf(c[r] - mrow[r]) * lrinv[r];
        int row = ((lane >> 4) << 3) + r;
        int col = half * 16 + (lane & 15);
        pt[row * 32 + col] = (__bf16)p;
      }
    }
    __syncthreads();  // single-wave WG: cheap
    v16bf pa = load_a(pt, 32);
#pragma unroll
    for (int j = 0; j < 4; ++j) {
      v16bf b = load_b(vt + j * 16, HD);
      o[j] = wmma_bf16(pa, b, o[j]);
    }
    __syncthreads();
  }

  int bb = bh >> 4;   // bh = b*NH + h
  int h  = bh & 15;
#pragma unroll
  for (int j = 0; j < 4; ++j) {
#pragma unroll
    for (int r = 0; r < 8; ++r) {
      int t = qt * 16 + ((lane >> 4) << 3) + r;
      int d = j * 16 + (lane & 15);
      Ab[((size_t)t * BATCH + bb) * EMB + h * HD + d] = (__bf16)o[j][r];
    }
  }
}

// ---------------------------------------------------------------------------
// Head-mean attention weights: out[b,t,s] = mean_h softmax(S_h)[t,s].
// Loops heads in-register -> deterministic, no atomics.
// grid = (T/16, T/16, B); one wave per block.
// ---------------------------------------------------------------------------
__global__ void mean_attn(const __bf16* __restrict__ Qb, const __bf16* __restrict__ Kb,
                          const float* __restrict__ mstat, const float* __restrict__ lstat,
                          float* __restrict__ outw) {
  int qt = blockIdx.x;
  int st = blockIdx.y;
  int bb = blockIdx.z;
  int lane = threadIdx.x & 31;

  float acc[8] = {};
  for (int h = 0; h < NH; ++h) {
    int bh = bb * NH + h;
    const __bf16* q  = Qb + ((size_t)bh * T_SEQ + qt * 16) * HD;
    const __bf16* kp = Kb + ((size_t)bh * T_SEQ + st * 16) * HD;
    v8f c = {};
    c = wmma_bf16(load_a(q, HD),      load_bT(kp, HD),      c);
    c = wmma_bf16(load_a(q + 32, HD), load_bT(kp + 32, HD), c);
#pragma unroll
    for (int r = 0; r < 8; ++r) {
      int t = qt * 16 + ((lane >> 4) << 3) + r;
      acc[r] += __expf(c[r] - mstat[(size_t)bh * T_SEQ + t]) /
                lstat[(size_t)bh * T_SEQ + t];
    }
  }
#pragma unroll
  for (int r = 0; r < 8; ++r) {
    int t = qt * 16 + ((lane >> 4) << 3) + r;
    int s = st * 16 + (lane & 15);
    outw[((size_t)bb * T_SEQ + t) * T_SEQ + s] = acc[r] * (1.0f / NH);
  }
}

// ---------------------------------------------------------------------------
// Output projection: out[m,f] = Ab[m,:] . Wo[:,f] + bias[f]  (fp32 out)
// One wave per block, 32x64 output tile.
// ---------------------------------------------------------------------------
__global__ void gemm_out(const __bf16* __restrict__ Ab, const __bf16* __restrict__ Wo,
                         const float* __restrict__ bias, float* __restrict__ out) {
  int mt = blockIdx.x;
  int nt = blockIdx.y;
  int lane = threadIdx.x & 31;

  v8f acc[2][4] = {};
  const __bf16* arow = Ab + (size_t)mt * 32 * EMB;
  for (int k0 = 0; k0 < EMB; k0 += 32) {
    if (k0 + 32 < EMB) {
      __builtin_prefetch(arow + k0 + 32, 0, 0);
      __builtin_prefetch(arow + 16 * EMB + k0 + 32, 0, 0);
    }
    v16bf a0 = load_a(arow + k0, EMB);
    v16bf a1 = load_a(arow + (size_t)16 * EMB + k0, EMB);
#pragma unroll
    for (int j = 0; j < 4; ++j) {
      v16bf b = load_b(Wo + (size_t)k0 * EMB + nt * 64 + j * 16, EMB);
      acc[0][j] = wmma_bf16(a0, b, acc[0][j]);
      acc[1][j] = wmma_bf16(a1, b, acc[1][j]);
    }
  }

#pragma unroll
  for (int u = 0; u < 2; ++u) {
#pragma unroll
    for (int j = 0; j < 4; ++j) {
      int f = nt * 64 + j * 16 + (lane & 15);
      float bv = bias[f];
#pragma unroll
      for (int r = 0; r < 8; ++r) {
        int m = mt * 32 + u * 16 + ((lane >> 4) << 3) + r;
        out[(size_t)m * EMB + f] = acc[u][j][r] + bv;
      }
    }
  }
}

// ---------------------------------------------------------------------------
// Host launch
// ---------------------------------------------------------------------------
extern "C" void kernel_launch(void* const* d_in, const int* in_sizes, int n_in,
                              void* d_out, int out_size, void* d_ws, size_t ws_size,
                              hipStream_t stream) {
  const float* query = (const float*)d_in[0];
  const float* w_in  = (const float*)d_in[1];
  const float* b_in  = (const float*)d_in[2];
  const float* w_out = (const float*)d_in[3];
  const float* b_out = (const float*)d_in[4];

  char* ws = (char*)d_ws;
  size_t off = 0;
  auto carve = [&](size_t bytes) -> char* {
    char* p = ws + off;
    off += (bytes + 255) & ~(size_t)255;
    return p;
  };

  __bf16* Xbf  = (__bf16*)carve((size_t)M_ROWS * EMB * 2);
  __bf16* Wbf  = (__bf16*)carve((size_t)N3 * EMB * 2);
  __bf16* Wobf = (__bf16*)carve((size_t)EMB * EMB * 2);
  __bf16* Qb   = (__bf16*)carve((size_t)BHEADS * T_SEQ * HD * 2);
  __bf16* Kb   = (__bf16*)carve((size_t)BHEADS * T_SEQ * HD * 2);
  __bf16* Vb   = (__bf16*)carve((size_t)BHEADS * T_SEQ * HD * 2);
  __bf16* Ab   = (__bf16*)carve((size_t)M_ROWS * EMB * 2);
  float*  mst  = (float*)carve((size_t)BHEADS * T_SEQ * 4);
  float*  lst  = (float*)carve((size_t)BHEADS * T_SEQ * 4);

  float* out_main = (float*)d_out;                       // [T, B, E]
  float* out_attn = out_main + (size_t)M_ROWS * EMB;     // [B, T, T]

  // 1) fp32 -> bf16 conversions
  {
    int n4 = (M_ROWS * EMB) / 4;
    cvt_f32_bf16<<<(n4 + 255) / 256, 256, 0, stream>>>(query, Xbf, n4);
    n4 = (N3 * EMB) / 4;
    cvt_f32_bf16<<<(n4 + 255) / 256, 256, 0, stream>>>(w_in, Wbf, n4);
    n4 = (EMB * EMB) / 4;
    cvt_f32_bf16<<<(n4 + 255) / 256, 256, 0, stream>>>(w_out, Wobf, n4);
  }

  // 2) QKV projection (32x64 tiles per wave)
  gemm_qkv<<<dim3(M_ROWS / 32, N3 / 64), 32, 0, stream>>>(Xbf, Wbf, b_in, Qb, Kb, Vb);

  // 3) softmax stats (online max/sumexp)
  attn_stats<<<dim3(T_SEQ / 16, BHEADS), 32, 0, stream>>>(Qb, Kb, mst, lst);

  // 4) P @ V with TDM-staged K/V tiles
  attn_av<<<dim3(T_SEQ / 16, BHEADS), 32, 10240, stream>>>(Qb, Kb, Vb, mst, lst, Ab);

  // 5) head-mean attention weights
  mean_attn<<<dim3(T_SEQ / 16, T_SEQ / 16, BATCH), 32, 0, stream>>>(Qb, Kb, mst, lst, out_attn);

  // 6) output projection (32x64 tiles per wave)
  gemm_out<<<dim3(M_ROWS / 32, EMB / 64), 32, 0, stream>>>(Ab, Wobf, b_out, out_main);
}
